// MHSA_75290776699428
// MI455X (gfx1250) — matrix operand order repaired and checked by reading, not compiled, not run on hardware
//
#include <hip/hip_runtime.h>
#include <hip/hip_bf16.h>

typedef __attribute__((ext_vector_type(16))) __bf16 v16bf;
typedef __attribute__((ext_vector_type(8)))  __bf16 v8bf;
typedef __attribute__((ext_vector_type(8)))  float  v8f;

union ABf { v16bf v; v8bf h[2]; };

constexpr int Bc  = 4;
constexpr int Sc  = 2048;
constexpr int Dc  = 1024;
constexpr int Hc  = 16;
constexpr int HDc = 64;
constexpr int Mtot = Bc * Sc;          // 8192 rows of x / ctx

__device__ __forceinline__ v8f wmma_bf16(v16bf a, v16bf b, v8f c) {
  // D = A(16x32 bf16) * B(32x16 bf16) + C(16x16 f32)
  return __builtin_amdgcn_wmma_f32_16x16x32_bf16(false, a, false, b, (short)0, c,
                                                 false, false);
}

// ---------------------------------------------------------------- f32 -> bf16
__global__ void cvt_f32_bf16(const float* __restrict__ in, __bf16* __restrict__ out,
                             int n) {
  for (int i = blockIdx.x * blockDim.x + threadIdx.x; i < n;
       i += gridDim.x * blockDim.x) {
    out[i] = (__bf16)in[i];
  }
}

// ------------------------------------------------------------------ WMMA GEMM
// C[M=8192][N=1024] = A[M][K=1024] (bf16 row-major) * W^T, with W = [N][K] bf16
// row-major ([out,in] as in nn.Linear). Block: 256 threads = 8 waves, each wave
// owns a 16-row strip of a 128x128 output tile (8 WMMA accumulators).
//
// Weight tile (128 cols x 32 K = 8KB) is staged in LDS via
// global_load_async_to_lds_b128 (ASYNCcnt), double-buffered: issue next tile,
// s_wait_asynccnt 2 on current, barrier, compute. This removes the 8x
// redundant per-wave global weight loads and hides global latency.
//
// mode 0: store f32 to outF[m*N + n]                         (final projection)
// mode 1: store bf16 to outB[((b*H+h)*S + s)*HD + d]         (Q, K layout)
// mode 2: store bf16 to outB[((b*H+h)*HD + d)*S + s]         (V transposed)
__global__ __launch_bounds__(256)
void gemm_wmma(const __bf16* __restrict__ A, const __bf16* __restrict__ W,
               float* __restrict__ outF, __bf16* __restrict__ outB, int mode) {
  __shared__ __bf16 wtile[2][128 * 32];   // [buf][n_local * 32 + k_local]

  const int tid  = threadIdx.x;
  const int wave = tid >> 5;
  const int lane = tid & 31;
  const int half = lane >> 4;
  const int lcol = lane & 15;
  const int mbase = blockIdx.x * 128 + wave * 16;
  const int nbase = blockIdx.y * 128;
  constexpr int K = Dc, N = Dc;

  // this thread's slice of the async weight-tile copy: 16 bf16 (two b128s)
  const int cn = tid >> 1;             // 0..127 : local column (n)
  const int ck = (tid & 1) * 16;       // 0 / 16 : local k offset
  const __bf16* wsrc0 = W + (size_t)(nbase + cn) * K + ck;

  auto issue_async = [&](int buf, int kk) {
    const __bf16* src = wsrc0 + kk;
    unsigned dst = (unsigned)(unsigned long long)&wtile[buf][cn * 32 + ck];
    asm volatile("global_load_async_to_lds_b128 %0, %1, off"
                 :: "v"(dst), "v"(src) : "memory");
    asm volatile("global_load_async_to_lds_b128 %0, %1, off offset:16"
                 :: "v"(dst), "v"(src) : "memory");
  };

  v8f acc[8];
#pragma unroll
  for (int f = 0; f < 8; ++f)
#pragma unroll
    for (int i = 0; i < 8; ++i) acc[f][i] = 0.f;

  // A-fragment rows: lane%16 within the wave's 16-row strip.
  const __bf16* arow = A + (size_t)(mbase + lcol) * K;

  issue_async(0, 0);
  for (int kk = 0; kk < K; kk += 32) {
    const int buf = (kk >> 5) & 1;
    if (kk + 32 < K) {
      issue_async(buf ^ 1, kk + 32);
      asm volatile("s_wait_asynccnt 0x2" ::: "memory");  // current buf landed
    } else {
      asm volatile("s_wait_asynccnt 0x0" ::: "memory");
    }
    __syncthreads();                   // all threads' async writes visible

    ABf a;
    // A layout: K = half*16 + e  (e = 0..15, 32B contiguous)
    a.h[0] = *(const v8bf*)(arow + kk + half * 16);
    a.h[1] = *(const v8bf*)(arow + kk + half * 16 + 8);
    if (kk + 32 < K) __builtin_prefetch((const void*)(arow + kk + 32), 0, 1);

    const __bf16* wt = wtile[buf];
#pragma unroll
    for (int f = 0; f < 8; ++f) {
      // B layout: col n = lcol, K = half*16 + e -> 32B contiguous from LDS
      v16bf bw = *(const v16bf*)(wt + (f * 16 + lcol) * 32 + half * 16);
      acc[f] = wmma_bf16(a.v, bw, acc[f]);
    }
    __syncthreads();                   // reads done before buf is overwritten
  }

  // C layout: row m = v + 8*half, col n = lcol
#pragma unroll
  for (int f = 0; f < 8; ++f) {
    const int n = nbase + f * 16 + lcol;
#pragma unroll
    for (int v = 0; v < 8; ++v) {
      const int m = mbase + v + 8 * half;
      const float val = acc[f][v];
      if (mode == 0) {
        outF[(size_t)m * N + n] = val;
      } else {
        const int b = m / Sc, s = m % Sc;
        const int h = n / HDc, d = n % HDc;
        if (mode == 1)
          outB[(((size_t)b * Hc + h) * Sc + s) * HDc + d] = (__bf16)val;
        else
          outB[(((size_t)b * Hc + h) * HDc + d) * Sc + s] = (__bf16)val;
      }
    }
  }
}

// ----------------------------------------------------------- flash attention
// One wave per 16-row query tile of one (b,h). Streams 32-key tiles:
// 4 WMMA for QK^T, online softmax (scale 1/8, additive -1e9 mask + post-softmax
// multiplicative mask as in the reference), P->LDS->A-fragment, 4 WMMA for PV.
__global__ __launch_bounds__(128)
void flash_attn(const __bf16* __restrict__ qb, const __bf16* __restrict__ kb,
                const __bf16* __restrict__ vtb, const unsigned char* __restrict__ mask,
                __bf16* __restrict__ ctx) {
  __shared__ __bf16 plds[4][16 * 32];   // per-wave P staging tile

  const int tid  = threadIdx.x;
  const int wave = tid >> 5;
  const int lane = tid & 31;
  const int half = lane >> 4;
  const int lcol = lane & 15;

  const int tile = blockIdx.x * 4 + wave;        // 8192 tiles total
  const int qt = tile & (Sc / 16 - 1);           // q-tile within (b,h)
  const int h  = (tile >> 7) & (Hc - 1);
  const int b  = tile >> 11;
  const int qbase = qt * 16;

  // ---- load Q fragments (16 rows x 64 dims -> two 16x32 A-fragments)
  const __bf16* qrow = qb + (((size_t)b * Hc + h) * Sc + qbase + lcol) * HDc;
  ABf aq[2];
#pragma unroll
  for (int j = 0; j < 2; ++j) {
    aq[j].h[0] = *(const v8bf*)(qrow + j * 32 + half * 16);
    aq[j].h[1] = *(const v8bf*)(qrow + j * 32 + half * 16 + 8);
  }

  v8f o[4];
#pragma unroll
  for (int f = 0; f < 4; ++f)
#pragma unroll
    for (int i = 0; i < 8; ++i) o[f][i] = 0.f;
  float rmax[8], rsum[8];
#pragma unroll
  for (int v = 0; v < 8; ++v) { rmax[v] = -3.0e38f; rsum[v] = 0.f; }

  const __bf16* kbase_p  = kb  + ((size_t)b * Hc + h) * Sc * HDc;
  const __bf16* vtbase_p = vtb + ((size_t)b * Hc + h) * HDc * (size_t)Sc;
  const unsigned char* mrow = mask + ((size_t)b * Sc + qbase) * Sc;

  for (int kk = 0; kk < Sc; kk += 32) {
    // ---- scores for two 16-key sub-tiles
    v8f sc[2];
#pragma unroll
    for (int t = 0; t < 2; ++t) {
      const __bf16* krow = kbase_p + (size_t)(kk + t * 16 + lcol) * HDc;
      v16bf bl = *(const v16bf*)(krow + half * 16);        // dims 0..31
      v16bf bh = *(const v16bf*)(krow + 32 + half * 16);   // dims 32..63
      v8f c;
#pragma unroll
      for (int i = 0; i < 8; ++i) c[i] = 0.f;
      c = wmma_bf16(aq[0].v, bl, c);
      c = wmma_bf16(aq[1].v, bh, c);
      sc[t] = c;
    }

    // ---- scale + masks
    float mv[2][8];
#pragma unroll
    for (int t = 0; t < 2; ++t) {
      const int key = kk + t * 16 + lcol;
#pragma unroll
      for (int v = 0; v < 8; ++v) {
        const int m = v + 8 * half;
        const unsigned char mk = mrow[(size_t)m * Sc + key];
        mv[t][v] = mk ? 1.f : 0.f;
        sc[t][v] = sc[t][v] * 0.125f + (mk ? 0.f : -1e9f);
      }
    }

    // ---- online softmax (row reductions across the 16 lanes of each half)
    float corr[8];
#pragma unroll
    for (int v = 0; v < 8; ++v) {
      float tmax = fmaxf(sc[0][v], sc[1][v]);
#pragma unroll
      for (int off = 8; off; off >>= 1)
        tmax = fmaxf(tmax, __shfl_xor(tmax, off, 16));
      const float nmax = fmaxf(rmax[v], tmax);
      corr[v] = __expf(rmax[v] - nmax);
      rmax[v] = nmax;
      float p0 = __expf(sc[0][v] - nmax) * mv[0][v];
      float p1 = __expf(sc[1][v] - nmax) * mv[1][v];
      sc[0][v] = p0; sc[1][v] = p1;
      float ps = p0 + p1;
#pragma unroll
      for (int off = 8; off; off >>= 1) ps += __shfl_xor(ps, off, 16);
      rsum[v] = rsum[v] * corr[v] + ps;
    }
#pragma unroll
    for (int f = 0; f < 4; ++f)
#pragma unroll
      for (int v = 0; v < 8; ++v) o[f][v] *= corr[v];

    // ---- P (C-layout f32) -> LDS (row-major bf16) -> A-fragment
    __bf16* my = plds[wave];
#pragma unroll
    for (int t = 0; t < 2; ++t)
#pragma unroll
      for (int v = 0; v < 8; ++v)
        my[(v + 8 * half) * 32 + t * 16 + lcol] = (__bf16)sc[t][v];
    __syncthreads();
    ABf pa;
    pa.h[0] = *(const v8bf*)(my + lcol * 32 + half * 16);
    pa.h[1] = *(const v8bf*)(my + lcol * 32 + half * 16 + 8);
    __syncthreads();

    // ---- PV: out[16 x 64] += P(16x32) x V(32x64), V pre-transposed
#pragma unroll
    for (int f = 0; f < 4; ++f) {
      const __bf16* vrow =
          vtbase_p + (size_t)(f * 16 + lcol) * Sc + kk + half * 16;
      v16bf vb = *(const v16bf*)vrow;
      o[f] = wmma_bf16(pa.v, vb, o[f]);
    }
  }

  // ---- normalize and emit ctx row-major [B*S][D] bf16
#pragma unroll
  for (int v = 0; v < 8; ++v) {
    const float inv = 1.f / fmaxf(rsum[v], 1e-30f);
    const int m = qbase + v + 8 * half;
#pragma unroll
    for (int f = 0; f < 4; ++f)
      ctx[((size_t)b * Sc + m) * Dc + h * HDc + f * 16 + lcol] =
          (__bf16)(o[f][v] * inv);
  }
}

// ------------------------------------------------------------------- launcher
extern "C" void kernel_launch(void* const* d_in, const int* in_sizes, int n_in,
                              void* d_out, int out_size, void* d_ws, size_t ws_size,
                              hipStream_t stream) {
  const float* x  = (const float*)d_in[0];
  const unsigned char* mask = (const unsigned char*)d_in[1];   // jax bool = 1B
  const float* wq = (const float*)d_in[2];
  const float* wk = (const float*)d_in[3];
  const float* wv = (const float*)d_in[4];
  const float* wo = (const float*)d_in[5];
  float* out = (float*)d_out;

  char* ws = (char*)d_ws;
  const size_t MB = 1ull << 20;
  __bf16* xb  = (__bf16*)(ws + 0 * MB);    // 16 MiB  8192x1024 bf16
  __bf16* wqB = (__bf16*)(ws + 16 * MB);   //  2 MiB each
  __bf16* wkB = (__bf16*)(ws + 18 * MB);
  __bf16* wvB = (__bf16*)(ws + 20 * MB);
  __bf16* woB = (__bf16*)(ws + 22 * MB);
  __bf16* qb  = (__bf16*)(ws + 24 * MB);   // 16 MiB  [B][H][S][HD]
  __bf16* kbv = (__bf16*)(ws + 40 * MB);   // 16 MiB  [B][H][S][HD]
  __bf16* vtb = (__bf16*)(ws + 56 * MB);   // 16 MiB  [B][H][HD][S]
  __bf16* ctx = (__bf16*)(ws + 72 * MB);   // 16 MiB  [B*S][D]

  const int nX = Mtot * Dc;                // 8,388,608
  const int nW = Dc * Dc;                  // 1,048,576
  cvt_f32_bf16<<<4096, 256, 0, stream>>>(x,  xb,  nX);
  cvt_f32_bf16<<<1024, 256, 0, stream>>>(wq, wqB, nW);
  cvt_f32_bf16<<<1024, 256, 0, stream>>>(wk, wkB, nW);
  cvt_f32_bf16<<<1024, 256, 0, stream>>>(wv, wvB, nW);
  cvt_f32_bf16<<<1024, 256, 0, stream>>>(wo, woB, nW);

  dim3 ggrid(Mtot / 128, Dc / 128);        // 64 x 8
  gemm_wmma<<<ggrid, 256, 0, stream>>>(xb, wqB, nullptr, qb,  1);
  gemm_wmma<<<ggrid, 256, 0, stream>>>(xb, wkB, nullptr, kbv, 1);
  gemm_wmma<<<ggrid, 256, 0, stream>>>(xb, wvB, nullptr, vtb, 2);

  const int ntiles = Bc * Hc * (Sc / 16);  // 8192 wave-tiles, 4 per block
  flash_attn<<<ntiles / 4, 128, 0, stream>>>(qb, kbv, vtb, mask, ctx);

  gemm_wmma<<<ggrid, 256, 0, stream>>>(ctx, woB, out, nullptr, 0);
}
